// Mamba2Mixer_40029095199172
// MI455X (gfx1250) — compile-verified, matrix operand order checked by reading, not run on hardware
//
#include <hip/hip_runtime.h>

// ---------------- constants ----------------
#define T_LEN   4096
#define D_MODEL 2048
#define D_INNER 4096
#define NHEADS  64
#define HEAD_DIM 64
#define D_STATE 128
#define D_IN_PROJ 8512   // 2*4096 + 2*128 + 64
#define CONV_DIM  4352   // 4096 + 2*128
#define NCHUNK  16
#define CHUNK   256
#define LDT     40       // LDS row stride (elements); 80 B = 5 x 16 B, keeps b128 alignment + bank spread

typedef __bf16 bf16_t;
typedef bf16_t v16bf __attribute__((ext_vector_type(16)));
typedef float  v8f   __attribute__((ext_vector_type(8)));

// ---------------- scalar helpers ----------------
__device__ __forceinline__ unsigned short f32_to_bf16(float f) {
    unsigned int u = __float_as_uint(f);
    unsigned int r = u + 0x7FFFu + ((u >> 16) & 1u);   // round-to-nearest-even
    return (unsigned short)(r >> 16);
}
__device__ __forceinline__ float bf16_to_f32(unsigned short h) {
    return __uint_as_float(((unsigned int)h) << 16);
}
__device__ __forceinline__ float silu_f(float x) {
    return x / (1.0f + __expf(-x));
}
// Generic shared-memory pointer -> 32-bit LDS offset (ISA: LDS aperture is in addr[63:32],
// addr[31:0] is the wave-relative LDS byte address).
__device__ __forceinline__ unsigned lds_addr32(const void* p) {
    return (unsigned)(unsigned long long)p;
}

// Fragment loader: 16x32 bf16 operand from LDS tile with row-stride LDT.
// ISA 16-bit A layout: lanes 0-15 = rows 0-15 holding K=0..7,16..23;
// lanes 16-31 = rows 0-15 holding K=8..15,24..31. B (as [N,K]) uses the mirror layout.
__device__ __forceinline__ v16bf load_frag(const unsigned short* smem, int rowBase, int lane) {
    const unsigned short* p = smem + (size_t)(rowBase + (lane & 15)) * LDT + ((lane >> 4) << 3);
    union { v16bf v; unsigned short u[16]; } f;
#pragma unroll
    for (int i = 0; i < 8; ++i) { f.u[i] = p[i]; f.u[i + 8] = p[16 + i]; }
    return f.v;
}

// ---------------- fp32 -> bf16 convert ----------------
__global__ void cvt_bf16_kernel(const float* __restrict__ in, unsigned short* __restrict__ out, long long n) {
    long long i = (long long)blockIdx.x * blockDim.x + threadIdx.x;
    long long stride = (long long)gridDim.x * blockDim.x;
    for (; i < n; i += stride) out[i] = f32_to_bf16(in[i]);
}

// ---------------- generic batched bf16 WMMA GEMM: C = A * B^T ----------------
// A: [M,K] bf16 row-major (lda), B: [N,K] bf16 row-major (ldb), C: [M,N] f32 (ldc)
// grid = (N/64, M/128, batch); block = 256 (8 waves); wave -> 32x32 via 2x2 WMMA tiles.
// Tile staging uses CDNA5 async global->LDS DMA (ASYNCcnt), no VGPR round-trip.
__global__ void gemm_bf16_wmma(const unsigned short* __restrict__ A,
                               const unsigned short* __restrict__ B,
                               float* __restrict__ C,
                               int M, int N, int K, int lda, int ldb, int ldc,
                               long long batchA, long long batchB, long long batchC) {
    A += (long long)blockIdx.z * batchA;
    B += (long long)blockIdx.z * batchB;
    C += (long long)blockIdx.z * batchC;
    const int tid = threadIdx.x, lane = tid & 31, w = tid >> 5;
    const int mBlock = blockIdx.y * 128, nBlock = blockIdx.x * 64;
    const int mw = (w >> 1) * 32, nw = (w & 1) * 32;

    __shared__ alignas(16) unsigned short sAm[128 * LDT];
    __shared__ alignas(16) unsigned short sBm[64 * LDT];

    v8f acc[2][2] = {};

    const int arow = tid >> 1, acb = (tid & 1) * 16;   // 128 rows x 32 cols, 16 elems (32 B)/thread
    const int brow = tid >> 2, bcb = (tid & 3) * 8;    // 64 rows x 32 cols, 8 elems (16 B)/thread
    const unsigned saA = lds_addr32(&sAm[arow * LDT + acb]);
    const unsigned saB = lds_addr32(&sBm[brow * LDT + bcb]);

    for (int kt = 0; kt < K; kt += 32) {
        {
            const unsigned short* gA = A + (size_t)(mBlock + arow) * lda + kt + acb;
            // 32 bytes per thread: two b128 async copies; offset applies to both LDS and global.
            asm volatile("global_load_async_to_lds_b128 %0, %1, off"
                         :: "v"(saA), "v"(gA) : "memory");
            asm volatile("global_load_async_to_lds_b128 %0, %1, off offset:16"
                         :: "v"(saA), "v"(gA) : "memory");
            const unsigned short* gB = B + (size_t)(nBlock + brow) * ldb + kt + bcb;
            asm volatile("global_load_async_to_lds_b128 %0, %1, off"
                         :: "v"(saB), "v"(gB) : "memory");
        }
        if (kt + 32 < K) {  // hint next K-slab toward the caches (global_prefetch_b8)
            __builtin_prefetch((const void*)(A + (size_t)(mBlock + arow) * lda + kt + 32 + acb), 0, 3);
            __builtin_prefetch((const void*)(B + (size_t)(nBlock + brow) * ldb + kt + 32 + bcb), 0, 3);
        }
        asm volatile("s_wait_asynccnt 0x0" ::: "memory");
        __syncthreads();
#pragma unroll
        for (int i = 0; i < 2; ++i) {
            v16bf a = load_frag(sAm, mw + i * 16, lane);
#pragma unroll
            for (int j = 0; j < 2; ++j) {
                v16bf b = load_frag(sBm, nw + j * 16, lane);
                acc[i][j] = __builtin_amdgcn_wmma_f32_16x16x32_bf16(
                    false, a, false, b, (short)0, acc[i][j], false, false);
            }
        }
        __syncthreads();
    }
    const int nL = lane & 15, mo = (lane >> 4) ? 8 : 0;
#pragma unroll
    for (int i = 0; i < 2; ++i)
#pragma unroll
        for (int j = 0; j < 2; ++j) {
            int mBase = mBlock + mw + i * 16 + mo;
            int nBase = nBlock + nw + j * 16 + nL;
#pragma unroll
            for (int r = 0; r < 8; ++r)
                C[(size_t)(mBase + r) * ldc + nBase] = acc[i][j][r];
        }
}

// ---------------- depthwise conv (D_CONV=4) + SiLU -> bf16 ----------------
__global__ void conv_silu_kernel(const float* __restrict__ zx,
                                 const float* __restrict__ conv_w,
                                 const float* __restrict__ conv_b,
                                 unsigned short* __restrict__ xbcb) {
    int t = blockIdx.x;
    int c = blockIdx.y * 256 + threadIdx.x;     // CONV_DIM = 17*256
    float acc = conv_b[c];
#pragma unroll
    for (int k = 0; k < 4; ++k) {
        int tt = t - 3 + k;
        if (tt >= 0) acc += conv_w[c * 4 + k] * zx[(size_t)tt * D_IN_PROJ + D_INNER + c];
    }
    xbcb[(size_t)t * CONV_DIM + c] = f32_to_bf16(silu_f(acc));
}

// ---------------- softplus(dt)+bias, dA, per-chunk inclusive cumsum ----------------
__global__ void dt_cumsum_kernel(const float* __restrict__ zx,
                                 const float* __restrict__ dt_bias,
                                 const float* __restrict__ A,
                                 float* __restrict__ dt_sp,
                                 float* __restrict__ dAcs) {
    int c = blockIdx.x, h = blockIdx.y, l = threadIdx.x;
    int t = c * CHUNK + l;
    float raw = zx[(size_t)t * D_IN_PROJ + (D_INNER + CONV_DIM) + h] + dt_bias[h];
    float sp = (raw > 20.0f) ? raw : log1pf(__expf(raw));
    dt_sp[t * NHEADS + h] = sp;
    __shared__ float sc[CHUNK];
    sc[l] = sp * A[h];
    __syncthreads();
    for (int o = 1; o < CHUNK; o <<= 1) {
        float add = (l >= o) ? sc[l - o] : 0.0f;
        __syncthreads();
        sc[l] += add;
        __syncthreads();
    }
    dAcs[t * NHEADS + h] = sc[l];
}

// ---------------- y_intra: scores = CB * L * dt (causal), y = scores @ x ----------------
// one block per (head, chunk); M=256(l) N=64(p) K=256(s) in 32-wide K steps
__global__ void y_intra_kernel(const unsigned short* __restrict__ xbcb,
                               const float* __restrict__ CB,
                               const float* __restrict__ dAcs,
                               const float* __restrict__ dt_sp,
                               float* __restrict__ y) {
    int h = blockIdx.x, c = blockIdx.y;
    int tid = threadIdx.x, lane = tid & 31, w = tid >> 5;
    __shared__ alignas(16) unsigned short sS[256 * LDT];   // scores tile [l][s-tile32]
    __shared__ alignas(16) unsigned short sX[64 * LDT];    // x^T tile   [p][s-tile32]
    v8f acc[2][4] = {};

    const int l = tid;
    const float dA_l = dAcs[(size_t)(c * CHUNK + l) * NHEADS + h];
    const float* CBrow = CB + ((size_t)c * CHUNK + l) * CHUNK;

    for (int ks = 0; ks < 8; ++ks) {
        int s0 = ks * 32;
#pragma unroll 4
        for (int s = 0; s < 32; ++s) {
            int sIdx = s0 + s;
            float v = 0.0f;
            if (sIdx <= l) {
                float dA_s = dAcs[(size_t)(c * CHUNK + sIdx) * NHEADS + h];
                float dts  = dt_sp[(size_t)(c * CHUNK + sIdx) * NHEADS + h];
                v = CBrow[sIdx] * __expf(dA_l - dA_s) * dts;
            }
            sS[l * LDT + s] = f32_to_bf16(v);
        }
        {
            int p = tid & 63, sb = (tid >> 6) * 8;
#pragma unroll
            for (int i = 0; i < 8; ++i) {
                int sIdx = s0 + sb + i;
                sX[p * LDT + sb + i] = xbcb[(size_t)(c * CHUNK + sIdx) * CONV_DIM + h * HEAD_DIM + p];
            }
        }
        __syncthreads();
#pragma unroll
        for (int i = 0; i < 2; ++i) {
            v16bf a = load_frag(sS, (w * 2 + i) * 16, lane);
#pragma unroll
            for (int j = 0; j < 4; ++j) {
                v16bf b = load_frag(sX, j * 16, lane);
                acc[i][j] = __builtin_amdgcn_wmma_f32_16x16x32_bf16(
                    false, a, false, b, (short)0, acc[i][j], false, false);
            }
        }
        __syncthreads();
    }
    const int nL = lane & 15, mo = (lane >> 4) ? 8 : 0;
#pragma unroll
    for (int i = 0; i < 2; ++i) {
        int mBase = (w * 2 + i) * 16 + mo;
#pragma unroll
        for (int j = 0; j < 4; ++j) {
            int nBase = j * 16 + nL;
#pragma unroll
            for (int r = 0; r < 8; ++r) {
                int tt = c * CHUNK + mBase + r;
                y[(size_t)tt * D_INNER + h * HEAD_DIM + nBase] = acc[i][j][r];
            }
        }
    }
}

// ---------------- states: states[c,h,p,n] = sum_l B[l,n] * xdt[l,p] ----------------
// M=64(p) N=128(n) K=256(l)
__global__ void states_kernel(const unsigned short* __restrict__ xbcb,
                              const float* __restrict__ dAcs,
                              const float* __restrict__ dt_sp,
                              float* __restrict__ states) {
    int h = blockIdx.x, c = blockIdx.y;
    int tid = threadIdx.x, lane = tid & 31, w = tid >> 5;
    __shared__ alignas(16) unsigned short sA[64 * LDT];    // xdt^T [p][l-tile32]
    __shared__ alignas(16) unsigned short sB[128 * LDT];   // B^T   [n][l-tile32]
    __shared__ float sScale[32];
    v8f acc[4] = {};
    const int mTile = (w & 3) * 16;
    const int nGrp  = (w >> 2) * 4;
    const float dA_last = dAcs[(size_t)(c * CHUNK + CHUNK - 1) * NHEADS + h];

    for (int kt = 0; kt < 8; ++kt) {
        int l0 = kt * 32;
        if (tid < 32) {
            int l = l0 + tid;
            float dA_l = dAcs[(size_t)(c * CHUNK + l) * NHEADS + h];
            sScale[tid] = dt_sp[(size_t)(c * CHUNK + l) * NHEADS + h] * __expf(dA_last - dA_l);
        }
        __syncthreads();
        {
            int p = tid & 63, lb = (tid >> 6) * 8;
#pragma unroll
            for (int i = 0; i < 8; ++i) {
                int l = lb + i;
                float xv = bf16_to_f32(xbcb[(size_t)(c * CHUNK + l0 + l) * CONV_DIM + h * HEAD_DIM + p]);
                sA[p * LDT + l] = f32_to_bf16(xv * sScale[l]);
            }
        }
        {
            int n = tid & 127, lb = (tid >> 7) * 16;
#pragma unroll
            for (int i = 0; i < 16; ++i) {
                int l = lb + i;
                sB[n * LDT + l] = xbcb[(size_t)(c * CHUNK + l0 + l) * CONV_DIM + D_INNER + n];
            }
        }
        __syncthreads();
        v16bf a = load_frag(sA, mTile, lane);
#pragma unroll
        for (int j = 0; j < 4; ++j) {
            v16bf b = load_frag(sB, (nGrp + j) * 16, lane);
            acc[j] = __builtin_amdgcn_wmma_f32_16x16x32_bf16(
                false, a, false, b, (short)0, acc[j], false, false);
        }
        __syncthreads();
    }
    const int nL = lane & 15, mo = (lane >> 4) ? 8 : 0;
    float* outp = states + ((size_t)(c * NHEADS + h)) * HEAD_DIM * D_STATE;
#pragma unroll
    for (int j = 0; j < 4; ++j) {
        int nBase = (nGrp + j) * 16 + nL;
#pragma unroll
        for (int r = 0; r < 8; ++r)
            outp[(size_t)(mTile + mo + r) * D_STATE + nBase] = acc[j][r];
    }
}

// ---------------- inter-chunk scan: S_prev[c] = S; S = cd*S + states[c] ----------------
__global__ void chunk_scan_kernel(const float* __restrict__ states,
                                  const float* __restrict__ dAcs,
                                  float* __restrict__ Sprev) {
    int h = blockIdx.x, tid = threadIdx.x;
    float S[32];
#pragma unroll
    for (int i = 0; i < 32; ++i) S[i] = 0.0f;
    for (int c = 0; c < NCHUNK; ++c) {
        float cd = __expf(dAcs[(size_t)(c * CHUNK + CHUNK - 1) * NHEADS + h]);
        size_t base = ((size_t)(c * NHEADS + h)) * (HEAD_DIM * D_STATE) + (size_t)tid * 32;
#pragma unroll
        for (int i = 0; i < 32; ++i) {
            Sprev[base + i] = S[i];
            S[i] = cd * S[i] + states[base + i];
        }
    }
}

// ---------------- y_inter + combine: y += (C @ Sprev^T)*exp(dAcs) + x*D ----------------
// M=256(l) N=64(p) K=128(n)
__global__ void y_inter_kernel(const unsigned short* __restrict__ xbcb,
                               const float* __restrict__ Sprev,
                               const float* __restrict__ dAcs,
                               const float* __restrict__ Dv,
                               float* __restrict__ y) {
    int h = blockIdx.x, c = blockIdx.y;
    int tid = threadIdx.x, lane = tid & 31, w = tid >> 5;
    __shared__ alignas(16) unsigned short sC[256 * LDT];   // C tile       [l][n-tile32]
    __shared__ alignas(16) unsigned short sB[64 * LDT];    // Sprev tile   [p][n-tile32] (f32->bf16)
    v8f acc[2][4] = {};

    for (int kt = 0; kt < 4; ++kt) {
        int n0 = kt * 32;
#pragma unroll 4
        for (int i = 0; i < 32; ++i)
            sC[tid * LDT + i] = xbcb[(size_t)(c * CHUNK + tid) * CONV_DIM + D_INNER + D_STATE + n0 + i];
        {
            int p = tid & 63, nb = (tid >> 6) * 8;
#pragma unroll
            for (int i = 0; i < 8; ++i) {
                float v = Sprev[((size_t)(c * NHEADS + h) * HEAD_DIM + p) * D_STATE + n0 + nb + i];
                sB[p * LDT + nb + i] = f32_to_bf16(v);
            }
        }
        __syncthreads();
#pragma unroll
        for (int i = 0; i < 2; ++i) {
            v16bf a = load_frag(sC, (w * 2 + i) * 16, lane);
#pragma unroll
            for (int j = 0; j < 4; ++j) {
                v16bf b = load_frag(sB, j * 16, lane);
                acc[i][j] = __builtin_amdgcn_wmma_f32_16x16x32_bf16(
                    false, a, false, b, (short)0, acc[i][j], false, false);
            }
        }
        __syncthreads();
    }
    const int nL = lane & 15, mo = (lane >> 4) ? 8 : 0;
    const float Dh = Dv[h];
#pragma unroll
    for (int i = 0; i < 2; ++i) {
        int mBase = (w * 2 + i) * 16 + mo;
#pragma unroll
        for (int j = 0; j < 4; ++j) {
            int nBase = j * 16 + nL;
#pragma unroll
            for (int r = 0; r < 8; ++r) {
                int tt = c * CHUNK + mBase + r;
                float dAl = dAcs[(size_t)tt * NHEADS + h];
                float xv = bf16_to_f32(xbcb[(size_t)tt * CONV_DIM + h * HEAD_DIM + nBase]);
                size_t yi = (size_t)tt * D_INNER + h * HEAD_DIM + nBase;
                y[yi] = y[yi] + acc[i][j][r] * __expf(dAl) + xv * Dh;
            }
        }
    }
}

// ---------------- gating (silu(z)) + RMSNorm -> bf16 ----------------
__global__ void gate_norm_kernel(const float* __restrict__ y,
                                 const float* __restrict__ zx,
                                 const float* __restrict__ norm_w,
                                 unsigned short* __restrict__ ynb) {
    int t = blockIdx.x, tid = threadIdx.x;
    __shared__ float red[256];
    float vals[16];
    float ss = 0.0f;
#pragma unroll
    for (int i = 0; i < 16; ++i) {
        int c = i * 256 + tid;
        float z = zx[(size_t)t * D_IN_PROJ + c];
        float v = y[(size_t)t * D_INNER + c] * silu_f(z);
        vals[i] = v;
        ss += v * v;
    }
    red[tid] = ss;
    __syncthreads();
    for (int o = 128; o > 0; o >>= 1) {
        if (tid < o) red[tid] += red[tid + o];
        __syncthreads();
    }
    float rs = rsqrtf(red[0] / (float)D_INNER + 1e-5f);
#pragma unroll
    for (int i = 0; i < 16; ++i) {
        int c = i * 256 + tid;
        ynb[(size_t)t * D_INNER + c] = f32_to_bf16(vals[i] * rs * norm_w[c]);
    }
}

// ---------------- host launcher ----------------
extern "C" void kernel_launch(void* const* d_in, const int* in_sizes, int n_in,
                              void* d_out, int out_size, void* d_ws, size_t ws_size,
                              hipStream_t stream) {
    const float* hs      = (const float*)d_in[0];
    const float* w_in    = (const float*)d_in[1];
    const float* conv_w  = (const float*)d_in[2];
    const float* conv_b  = (const float*)d_in[3];
    const float* Av      = (const float*)d_in[4];
    const float* Dvp     = (const float*)d_in[5];
    const float* dt_bias = (const float*)d_in[6];
    const float* norm_w  = (const float*)d_in[7];
    const float* w_out   = (const float*)d_in[8];
    float* out = (float*)d_out;

    char* ws = (char*)d_ws;
    size_t off = 0;
    auto alloc = [&](size_t bytes) -> void* {
        void* p = ws + off;
        off += (bytes + 255) & ~(size_t)255;
        return p;
    };
    unsigned short* hsb   = (unsigned short*)alloc((size_t)T_LEN * D_MODEL * 2);
    unsigned short* winb  = (unsigned short*)alloc((size_t)D_IN_PROJ * D_MODEL * 2);
    unsigned short* woutb = (unsigned short*)alloc((size_t)D_MODEL * D_INNER * 2);
    float* zx             = (float*)alloc((size_t)T_LEN * D_IN_PROJ * 4);
    unsigned short* xbcb  = (unsigned short*)alloc((size_t)T_LEN * CONV_DIM * 2);
    float* dt_sp          = (float*)alloc((size_t)T_LEN * NHEADS * 4);
    float* dAcs           = (float*)alloc((size_t)T_LEN * NHEADS * 4);
    float* CBbuf          = (float*)alloc((size_t)NCHUNK * CHUNK * CHUNK * 4);
    float* statesB        = (float*)alloc((size_t)NCHUNK * NHEADS * HEAD_DIM * D_STATE * 4);
    float* SprevB         = (float*)alloc((size_t)NCHUNK * NHEADS * HEAD_DIM * D_STATE * 4);
    float* ybuf           = (float*)alloc((size_t)T_LEN * D_INNER * 4);
    unsigned short* ynb   = (unsigned short*)alloc((size_t)T_LEN * D_INNER * 2);

    // 1) fp32 -> bf16 conversions
    cvt_bf16_kernel<<<dim3(2048), 256, 0, stream>>>(hs, hsb, (long long)T_LEN * D_MODEL);
    cvt_bf16_kernel<<<dim3(4096), 256, 0, stream>>>(w_in, winb, (long long)D_IN_PROJ * D_MODEL);
    cvt_bf16_kernel<<<dim3(2048), 256, 0, stream>>>(w_out, woutb, (long long)D_MODEL * D_INNER);

    // 2) in_proj GEMM: zx = hs @ w_in^T   [4096 x 8512]
    gemm_bf16_wmma<<<dim3(D_IN_PROJ / 64, T_LEN / 128, 1), 256, 0, stream>>>(
        hsb, winb, zx, T_LEN, D_IN_PROJ, D_MODEL, D_MODEL, D_MODEL, D_IN_PROJ, 0, 0, 0);

    // 3) conv + SiLU -> xbc (bf16)
    conv_silu_kernel<<<dim3(T_LEN, CONV_DIM / 256), 256, 0, stream>>>(zx, conv_w, conv_b, xbcb);

    // 4) softplus(dt)+cumsum per chunk/head
    dt_cumsum_kernel<<<dim3(NCHUNK, NHEADS), CHUNK, 0, stream>>>(zx, dt_bias, Av, dt_sp, dAcs);

    // 5) CB = Cc @ Bc^T per chunk  [256 x 256, K=128]
    gemm_bf16_wmma<<<dim3(CHUNK / 64, CHUNK / 128, NCHUNK), 256, 0, stream>>>(
        xbcb + D_INNER + D_STATE, xbcb + D_INNER, CBbuf,
        CHUNK, CHUNK, D_STATE, CONV_DIM, CONV_DIM, CHUNK,
        (long long)CHUNK * CONV_DIM, (long long)CHUNK * CONV_DIM,
        (long long)CHUNK * CHUNK);

    // 6) intra-chunk attention-like output
    y_intra_kernel<<<dim3(NHEADS, NCHUNK), 256, 0, stream>>>(xbcb, CBbuf, dAcs, dt_sp, ybuf);

    // 7) per-chunk states
    states_kernel<<<dim3(NHEADS, NCHUNK), 256, 0, stream>>>(xbcb, dAcs, dt_sp, statesB);

    // 8) inter-chunk sequential scan
    chunk_scan_kernel<<<dim3(NHEADS), 256, 0, stream>>>(statesB, dAcs, SprevB);

    // 9) inter-chunk output + D skip term, combined into ybuf
    y_inter_kernel<<<dim3(NHEADS, NCHUNK), 256, 0, stream>>>(xbcb, SprevB, dAcs, Dvp, ybuf);

    // 10) gating + RMSNorm -> bf16
    gate_norm_kernel<<<dim3(T_LEN), 256, 0, stream>>>(ybuf, zx, norm_w, ynb);

    // 11) out_proj GEMM: out = ynorm @ w_out^T  [4096 x 2048]
    gemm_bf16_wmma<<<dim3(D_MODEL / 64, T_LEN / 128, 1), 256, 0, stream>>>(
        ynb, woutb, out, T_LEN, D_MODEL, D_INNER, D_INNER, D_INNER, D_MODEL, 0, 0, 0);
}
